// SE3EquivariantLayer_34187939676350
// MI455X (gfx1250) — matrix operand order repaired and checked by reading, not compiled, not run on hardware
//
#include <hip/hip_runtime.h>

typedef __attribute__((ext_vector_type(16))) _Float16 v16h;
typedef __attribute__((ext_vector_type(8)))  float    v8f;

#define LL       2048
#define EDGE_DIM 32
#define NODE_DIM 64
#define NWAVES   8
#define TILES    (LL / 16)            // 128 tiles of 16 j's per row
#define ITERS    (TILES / NWAVES)     // 16 tiles per wave, uniform trip count

__device__ __forceinline__ void sm_merge(float& m, float& s, float& vx, float& vy, float& vz,
                                         float m2, float s2, float x2, float y2, float z2) {
    float M  = fmaxf(m, m2);
    float e1 = __expf(m  - M);
    float e2 = __expf(m2 - M);
    m = M;
    s  = s  * e1 + s2 * e2;
    vx = vx * e1 + x2 * e2;
    vy = vy * e1 + y2 * e2;
    vz = vz * e1 + z2 * e2;
}

__global__ __launch_bounds__(NWAVES * 32)
void se3_layer_kernel(const float* __restrict__ coords,
                      const float* __restrict__ node_features,
                      const float* __restrict__ pair_features,
                      const float* __restrict__ w_attn,   // 33 floats
                      const float* __restrict__ b_attn,   // 1
                      const float* __restrict__ w_mag,    // 64
                      const float* __restrict__ b_mag,    // 1
                      float* __restrict__ out) {
    __shared__ float scx[LL], scy[LL], scz[LL];
    __shared__ float red[NWAVES][8];

    const int i    = blockIdx.x;
    const int tid  = threadIdx.x;
    const int lane = tid & 31;
    const int wave = tid >> 5;
    const int l15  = lane & 15;

    // Stage coords into LDS (24 KB) — read once, reused 2048x per block.
    for (int idx = tid; idx < LL; idx += NWAVES * 32) {
        scx[idx] = coords[idx * 3 + 0];
        scy[idx] = coords[idx * 3 + 1];
        scz[idx] = coords[idx * 3 + 2];
    }
    __syncthreads();

    const float cix = scx[i], ciy = scy[i], ciz = scz[i];
    const float wd  = w_attn[EDGE_DIM];   // distance weight (w_attn[-1])
    const float ba  = b_attn[0];

    // A operand: w_attn broadcast to all 16 rows. Contiguous K layout (matches B):
    // lanes 0-15 hold K = 0..15 (h0..h15);
    // lanes 16-31 hold K = 16..31. Row index M is irrelevant (broadcast).
    v16h aW;
    {
        const int b0 = (lane < 16) ? 0 : 16;
#pragma unroll
        for (int t = 0; t < 8; ++t) {
            aW[t]     = (_Float16)w_attn[b0 + t];
            aW[8 + t] = (_Float16)w_attn[b0 + 8 + t];
        }
    }

    // Online-softmax state; lane L tracks j = tile*16 + (L&15).
    float m = -INFINITY, s = 0.0f, vx = 0.0f, vy = 0.0f, vz = 0.0f;

    const size_t rowbase = (size_t)i * LL * EDGE_DIM;
    const int    eoff    = (lane < 16) ? 0 : 16;   // B layout: half-wave owns K-halves

    // Base pointer for this wave's first tile; advance by a fixed stride so the
    // loop has a uniform trip count (no exec-mask churn around the WMMA).
    const float* rec = pair_features + rowbase
                     + (size_t)(wave * 16 + l15) * EDGE_DIM + eoff;
    int jj = wave * 16 + l15;

    for (int it = 0; it < ITERS; ++it) {
        // Unconditional speculative prefetch of this wave's next tile
        // (CDNA5 global_prefetch_b8; dropped silently if past the buffer).
        __builtin_prefetch(rec + NWAVES * 16 * EDGE_DIM, 0, 1);

        const float4 p0 = *(const float4*)(rec + 0);
        const float4 p1 = *(const float4*)(rec + 4);
        const float4 p2 = *(const float4*)(rec + 8);
        const float4 p3 = *(const float4*)(rec + 12);

        // B operand: 32x16 f16, column n = pair_features[i, tile*16+n, :].
        v16h bM;
        bM[0]  = (_Float16)p0.x; bM[1]  = (_Float16)p0.y;
        bM[2]  = (_Float16)p0.z; bM[3]  = (_Float16)p0.w;
        bM[4]  = (_Float16)p1.x; bM[5]  = (_Float16)p1.y;
        bM[6]  = (_Float16)p1.z; bM[7]  = (_Float16)p1.w;
        bM[8]  = (_Float16)p2.x; bM[9]  = (_Float16)p2.y;
        bM[10] = (_Float16)p2.z; bM[11] = (_Float16)p2.w;
        bM[12] = (_Float16)p3.x; bM[13] = (_Float16)p3.y;
        bM[14] = (_Float16)p3.z; bM[15] = (_Float16)p3.w;

        v8f c = {};
        c = __builtin_amdgcn_wmma_f32_16x16x32_f16(false, aW, false, bM,
                                                   (short)0, c, false, false);
        const float pf = c[0];   // dot(w_attn[0:32], pair[i, jj, :]) in every row slot

        // Distance / direction for this lane's j.
        const float dx = cix - scx[jj];
        const float dy = ciy - scy[jj];
        const float dz = ciz - scz[jj];
        const float d2   = fmaf(dx, dx, fmaf(dy, dy, dz * dz));
        const float dist = __builtin_amdgcn_sqrtf(d2) + 1e-8f;
        const float invd = __builtin_amdgcn_rcpf(dist);

        const float logit = fmaf(dist, wd, pf) + ba;

        // Online softmax update.
        const float Mn = fmaxf(m, logit);
        const float e1 = __expf(m - Mn);
        const float p  = __expf(logit - Mn);
        m  = Mn;
        s  = fmaf(s,  e1, p);
        vx = fmaf(vx, e1, p * dx * invd);
        vy = fmaf(vy, e1, p * dy * invd);
        vz = fmaf(vz, e1, p * dz * invd);

        rec += NWAVES * 16 * EDGE_DIM;   // next tile for this wave
        jj  += NWAVES * 16;
    }

    // Merge the 16 distinct lane states (half-waves hold duplicates; xor-merge is idempotent).
#pragma unroll
    for (int off = 8; off >= 1; off >>= 1) {
        float m2 = __shfl_xor(m,  off, 32);
        float s2 = __shfl_xor(s,  off, 32);
        float x2 = __shfl_xor(vx, off, 32);
        float y2 = __shfl_xor(vy, off, 32);
        float z2 = __shfl_xor(vz, off, 32);
        sm_merge(m, s, vx, vy, vz, m2, s2, x2, y2, z2);
    }

    if (lane == 0) {
        red[wave][0] = m;  red[wave][1] = s;
        red[wave][2] = vx; red[wave][3] = vy; red[wave][4] = vz;
    }
    __syncthreads();

    if (tid == 0) {
        float M  = red[0][0], S  = red[0][1];
        float Vx = red[0][2], Vy = red[0][3], Vz = red[0][4];
#pragma unroll
        for (int w = 1; w < NWAVES; ++w)
            sm_merge(M, S, Vx, Vy, Vz,
                     red[w][0], red[w][1], red[w][2], red[w][3], red[w][4]);

        // update magnitude: tanh(node_features[i] . w_mag + b_mag) * 0.1
        float acc = b_mag[0];
        const float* nf = node_features + (size_t)i * NODE_DIM;
#pragma unroll 4
        for (int e = 0; e < NODE_DIM; ++e)
            acc = fmaf(nf[e], w_mag[e], acc);
        const float mag  = tanhf(acc) * 0.1f;
        const float invS = __builtin_amdgcn_rcpf(S);

        out[i * 3 + 0] = cix + Vx * invS * mag;
        out[i * 3 + 1] = ciy + Vy * invS * mag;
        out[i * 3 + 2] = ciz + Vz * invS * mag;
    }
}

extern "C" void kernel_launch(void* const* d_in, const int* in_sizes, int n_in,
                              void* d_out, int out_size, void* d_ws, size_t ws_size,
                              hipStream_t stream) {
    const float* coords        = (const float*)d_in[0];
    const float* node_features = (const float*)d_in[1];
    const float* pair_features = (const float*)d_in[2];
    const float* w_attn        = (const float*)d_in[3];
    const float* b_attn        = (const float*)d_in[4];
    const float* w_mag         = (const float*)d_in[5];
    const float* b_mag         = (const float*)d_in[6];
    float*       out           = (float*)d_out;

    se3_layer_kernel<<<LL, NWAVES * 32, 0, stream>>>(
        coords, node_features, pair_features,
        w_attn, b_attn, w_mag, b_mag, out);
}
